// CfC_14456859918828
// MI455X (gfx1250) — matrix operand order repaired and checked by reading, not compiled, not run on hardware
//
#include <hip/hip_runtime.h>
#include <math.h>

// ---------------------------------------------------------------------------
// CfC scan for MI455X (gfx1250, wave32, WMMA 16x16x32 f16 -> f32 accum)
//
// B=256 batch rows -> 16 independent workgroups (16-row WMMA M-tile each),
// each running the full T=1024 recurrence. Weights live permanently in VGPRs
// as WMMA B-fragments; activations hand off through LDS; x is streamed two
// steps ahead with global_load_async_to_lds_b64 (ASYNCcnt). A-fragment LDS
// loads are ping-pong software-pipelined so ds_load latency hides behind the
// previous WMMA instead of serializing (no s_wait_dscnt 0 before each WMMA).
// ---------------------------------------------------------------------------

#define B_   256
#define T_   1024
#define I_   64
#define U_   128
#define BB_  256
#define HN_  512          // 4 * U_ (ff1 | ff2 | t_a | t_b fused)

#define INA_STRIDE 200    // 192 + 8 pad (halfs); 400B row stride, 16B aligned
#define ZA_STRIDE  264    // 256 + 8 pad (halfs); 528B row stride, 16B aligned

#define NWAVES   16
#define NTHREADS (NWAVES * 32)

typedef _Float16 v16h __attribute__((ext_vector_type(16)));
typedef _Float16 v8h  __attribute__((ext_vector_type(8)));
typedef float    v8f  __attribute__((ext_vector_type(8)));

// ---- branch-free transcendentals on the raw HW ops ------------------------
__device__ __forceinline__ float fast_tanh(float x) {
  float ax = fabsf(x);
  float e  = __builtin_amdgcn_exp2f(-2.8853900817779268f * ax);  // exp(-2|x|)
  float r  = (1.0f - e) * __builtin_amdgcn_rcpf(1.0f + e);
  return copysignf(r, x);
}
__device__ __forceinline__ float fast_sigmoid(float x) {
  return __builtin_amdgcn_rcpf(1.0f + __builtin_amdgcn_exp2f(-1.4426950408889634f * x));
}
__device__ __forceinline__ float lecun_tanh(float x) {
  return 1.7159f * fast_tanh(0.666f * x);
}

// Load a 16x32 f16 A-fragment (v16h per lane) from a row-major LDS tile.
// Lane L -> row M = L%16, half h = L/16.
//   elements 0..7  : K = ktbase      + 8*h + (0..7)   (16B contiguous)
//   elements 8..15 : K = ktbase + 16 + 8*h + (0..7)   (16B contiguous)
__device__ __forceinline__ v16h load_a_frag(const _Float16* lds, int strideHalfs,
                                            int m, int ktbase, int lh) {
  const _Float16* p = lds + m * strideHalfs + ktbase + 8 * lh;
  v8h lo = *(const v8h*)(p);
  v8h hi = *(const v8h*)(p + 16);
  return __builtin_shufflevector(lo, hi, 0, 1, 2, 3, 4, 5, 6, 7,
                                 8, 9, 10, 11, 12, 13, 14, 15);
}

// Gather a 32x16 f16 B-fragment from a row-major f32 weight [K x ldw].
// Lane L -> col N = L%16, half h = L/16; element e -> K = ktbase + 8*h + (e&7) + 16*(e>>3).
__device__ __forceinline__ v16h gather_b_frag(const float* W, int ldw,
                                              int ktbase, int col, int lh) {
  v16h f;
  const float* p = W + (size_t)(ktbase + 8 * lh) * ldw + col;
#pragma unroll
  for (int e = 0; e < 16; ++e) f[e] = (_Float16)p[(size_t)((e & 7) + 16 * (e >> 3)) * ldw];
  return f;
}

__device__ __forceinline__ v8f wmma_f16(v16h a, v16h b, v8f c) {
  return __builtin_amdgcn_wmma_f32_16x16x32_f16(false, a, false, b,
                                                (short)0, c, false, false);
}

// Async copy of 8 bytes (global -> LDS), tracked by ASYNCcnt.
__device__ __forceinline__ void async_copy_b64(unsigned ldsaddr, const float* gptr) {
  asm volatile("global_load_async_to_lds_b64 %0, %1, off"
               :: "v"(ldsaddr), "v"(gptr) : "memory");
}

__global__ __launch_bounds__(NTHREADS)
void cfc_scan_kernel(const float* __restrict__ x,
                     const float* __restrict__ Wbb,  const float* __restrict__ bbb,
                     const float* __restrict__ Wff1, const float* __restrict__ bff1,
                     const float* __restrict__ Wff2, const float* __restrict__ bff2,
                     const float* __restrict__ Wta,  const float* __restrict__ bta,
                     const float* __restrict__ Wtb,  const float* __restrict__ btb,
                     float* __restrict__ out) {
  // LDS handoff buffers (~61 KB of the 320 KB WGP pool)
  __shared__ __align__(16) _Float16 inA[2][16 * INA_STRIDE]; // [x_t | h] A staging, double buffered
  __shared__ __align__(16) _Float16 zA[16 * ZA_STRIDE];      // backbone output (f16 A for GEMM2)
  __shared__ __align__(16) float    headOut[16 * HN_];       // raw head accumulators
  __shared__ __align__(16) float    xstage[2][16 * I_];      // async f32 x staging (2 steps ahead)

  const int tid  = threadIdx.x;
  const int wave = tid >> 5;
  const int lane = tid & 31;
  const int lm   = lane & 15;   // N col (B/C/D) or M row (A)
  const int lh   = lane >> 4;   // lane half
  const int m0   = blockIdx.x * 16;

  // per-thread x staging coordinates (each thread owns the same 8B slot it loads)
  const int xj   = tid * 2;
  const int xrow = xj >> 6;
  const int xcol = xj & 63;
  const float* xbase = x + (size_t)(m0 + xrow) * T_ * I_ + xcol;

  // ------------------- one-time init: persistent weight fragments ----------
  const int zcol = wave * 16 + lm;
  v16h Bbb[6];
#pragma unroll
  for (int kt = 0; kt < 6; ++kt) Bbb[kt] = gather_b_frag(Wbb, BB_, kt * 32, zcol, lh);

  const float* Wh[4] = {Wff1, Wff2, Wta, Wtb};
  const float* bh[4] = {bff1, bff2, bta, btb};
  const int ca = wave * 32 + lm;        // head col of tile A
  const int cb = ca + 16;               // head col of tile B
  const float* WA = Wh[ca >> 7]; const int colA = ca & 127;
  const float* WB = Wh[cb >> 7]; const int colB = cb & 127;
  const float biasA = bh[ca >> 7][colA];
  const float biasB = bh[cb >> 7][colB];
  const float biasZ = bbb[zcol];

  v16h Bha[8], Bhb[8];
#pragma unroll
  for (int kt = 0; kt < 8; ++kt) {
    Bha[kt] = gather_b_frag(WA, U_, kt * 32, colA, lh);
    Bhb[kt] = gather_b_frag(WB, U_, kt * 32, colB, lh);
  }

  // ------------------- init state: h = 0, stage x_0 / x_1 ------------------
  {
    int hrow = tid >> 5, u4 = (tid & 31) * 4;
    uint2 z2; z2.x = 0u; z2.y = 0u;
    *(uint2*)&inA[0][hrow * INA_STRIDE + 64 + u4] = z2;    // h0 = 0
    float2 xv = *(const float2*)xbase;                     // x_0 direct
    _Float16* dst = &inA[0][xrow * INA_STRIDE + xcol];
    dst[0] = (_Float16)xv.x; dst[1] = (_Float16)xv.y;
    // x_1 -> xstage[1] via async DMA (consumed during step 0)
    async_copy_b64((unsigned)(size_t)&xstage[1][xj], xbase + I_);
  }
  __syncthreads();

  // ------------------- the scan -------------------------------------------
  for (int t = 0; t < T_; ++t) {
    const _Float16* curA = inA[t & 1];
    _Float16*       nxtA = inA[(t + 1) & 1];

    // kick off x_{t+2} -> xstage[t&1] (that buffer was consumed at step t-1)
    if (t + 2 < T_) {
      const float* gp = xbase + (size_t)(t + 2) * I_;
      async_copy_b64((unsigned)(size_t)&xstage[t & 1][xj], gp);
      if (t + 4 < T_) __builtin_prefetch(gp + 2 * I_, 0, 0);   // global_prefetch_b8
    }

    // ---- GEMM1: z = lecun_tanh([x_t | h] @ Wbb + b_bb) ----
    // ping-pong pipelined A-fragment loads (even/odd in distinct registers)
    v8f acc = {biasZ, biasZ, biasZ, biasZ, biasZ, biasZ, biasZ, biasZ};
    {
      v16h aEven = load_a_frag(curA, INA_STRIDE, lm, 0, lh);
      v16h aOdd  = load_a_frag(curA, INA_STRIDE, lm, 32, lh);
#pragma unroll
      for (int kt = 0; kt < 6; kt += 2) {
        acc = wmma_f16(aEven, Bbb[kt], acc);
        if (kt + 2 < 6) aEven = load_a_frag(curA, INA_STRIDE, lm, (kt + 2) * 32, lh);
        acc = wmma_f16(aOdd, Bbb[kt + 1], acc);
        if (kt + 3 < 6) aOdd = load_a_frag(curA, INA_STRIDE, lm, (kt + 3) * 32, lh);
      }
    }
#pragma unroll
    for (int r = 0; r < 8; ++r) {
      int row = r + 8 * lh;                       // C/D layout: VGPR r -> M = r + 8h
      zA[row * ZA_STRIDE + zcol] = (_Float16)lecun_tanh(acc[r]);
    }

    // ---- convert staged x_{t+1} (f32 LDS -> f16 A buffer), latency hidden ----
    if (t + 1 < T_) {
      if (t + 2 < T_) asm volatile("s_wait_asynccnt 0x1" ::: "memory");
      else            asm volatile("s_wait_asynccnt 0x0" ::: "memory");
      float2 xv = *(const float2*)&xstage[(t + 1) & 1][xj];
      _Float16* dst = &nxtA[xrow * INA_STRIDE + xcol];
      dst[0] = (_Float16)xv.x; dst[1] = (_Float16)xv.y;
    }
    __syncthreads();

    // ---- GEMM2: [ff1 | ff2 | t_a | t_b] = z @ Whead + bhead ----
    v8f acA = {biasA, biasA, biasA, biasA, biasA, biasA, biasA, biasA};
    v8f acB = {biasB, biasB, biasB, biasB, biasB, biasB, biasB, biasB};
    {
      v16h aEven = load_a_frag(zA, ZA_STRIDE, lm, 0, lh);
      v16h aOdd  = load_a_frag(zA, ZA_STRIDE, lm, 32, lh);
#pragma unroll
      for (int kt = 0; kt < 8; kt += 2) {
        acA = wmma_f16(aEven, Bha[kt], acA);
        acB = wmma_f16(aEven, Bhb[kt], acB);
        if (kt + 2 < 8) aEven = load_a_frag(zA, ZA_STRIDE, lm, (kt + 2) * 32, lh);
        acA = wmma_f16(aOdd, Bha[kt + 1], acA);
        acB = wmma_f16(aOdd, Bhb[kt + 1], acB);
        if (kt + 3 < 8) aOdd = load_a_frag(zA, ZA_STRIDE, lm, (kt + 3) * 32, lh);
      }
    }
#pragma unroll
    for (int r = 0; r < 8; ++r) {
      int row = r + 8 * lh;
      headOut[row * HN_ + ca] = acA[r];
      headOut[row * HN_ + cb] = acB[r];
    }
    __syncthreads();

    // ---- combine: h = ff1 + s*(ff2-ff1), s = sigmoid(t_a + t_b) ----
    {
      int row = tid >> 5;
      int u   = (tid & 31) * 4;
      const float* ho = &headOut[row * HN_];
      float4 f1 = *(const float4*)(ho + u);
      float4 f2 = *(const float4*)(ho + 128 + u);
      float4 a4 = *(const float4*)(ho + 256 + u);
      float4 b4 = *(const float4*)(ho + 384 + u);
      float hn[4];
      const float* pf1 = (const float*)&f1; const float* pf2 = (const float*)&f2;
      const float* pa4 = (const float*)&a4; const float* pb4 = (const float*)&b4;
#pragma unroll
      for (int j = 0; j < 4; ++j) {
        float ff1 = fast_tanh(pf1[j]);
        float ff2 = fast_tanh(pf2[j]);
        float s   = fast_sigmoid(pa4[j] + pb4[j]);
        hn[j] = ff1 + s * (ff2 - ff1);
      }
      float4 o; o.x = hn[0]; o.y = hn[1]; o.z = hn[2]; o.w = hn[3];
      *(float4*)(out + (size_t)(m0 + row) * T_ * U_ + (size_t)t * U_ + u) = o;   // readout[b,t,u]
      _Float16* hd = &nxtA[row * INA_STRIDE + 64 + u];                           // carry h (f16)
      hd[0] = (_Float16)hn[0]; hd[1] = (_Float16)hn[1];
      hd[2] = (_Float16)hn[2]; hd[3] = (_Float16)hn[3];
      if (t == T_ - 1) {
        *(float4*)(out + (size_t)B_ * T_ * U_ + (size_t)(m0 + row) * U_ + u) = o; // h_last
      }
    }
    __syncthreads();
  }
}

extern "C" void kernel_launch(void* const* d_in, const int* in_sizes, int n_in,
                              void* d_out, int out_size, void* d_ws, size_t ws_size,
                              hipStream_t stream) {
  (void)in_sizes; (void)n_in; (void)d_ws; (void)ws_size; (void)out_size;
  const float* x    = (const float*)d_in[0];
  const float* Wbb  = (const float*)d_in[1];
  const float* bbb  = (const float*)d_in[2];
  const float* Wff1 = (const float*)d_in[3];
  const float* bff1 = (const float*)d_in[4];
  const float* Wff2 = (const float*)d_in[5];
  const float* bff2 = (const float*)d_in[6];
  const float* Wta  = (const float*)d_in[7];
  const float* bta  = (const float*)d_in[8];
  const float* Wtb  = (const float*)d_in[9];
  const float* btb  = (const float*)d_in[10];
  float* out = (float*)d_out;

  dim3 grid(B_ / 16);
  dim3 block(NTHREADS);
  cfc_scan_kernel<<<grid, block, 0, stream>>>(x, Wbb, bbb, Wff1, bff1,
                                              Wff2, bff2, Wta, bta, Wtb, btb, out);
}
